// MultiHeadCrossAttention_37125697306956
// MI455X (gfx1250) — compile-verified, matrix operand order checked by reading, not run on hardware
//
#include <hip/hip_runtime.h>
#include <math.h>

// ---------------------------------------------------------------------------
// MultiHeadCrossAttention for MI455X (gfx1250, wave32, WMMA + TDM)
// B=8, N=1024, H=8, n=128 per head, scale=960, Cq in {64,128,256,512}
// GEMMs: v_wmma_f32_16x16x32_f16; A/B panels staged via tensor_load_to_lds
// (Tensor Data Mover) for non-transposed operands.
// ---------------------------------------------------------------------------

typedef __attribute__((ext_vector_type(16))) _Float16 v16h;
typedef __attribute__((ext_vector_type(8)))  float    v8f;
typedef __attribute__((ext_vector_type(4)))  float    v4f;
typedef __attribute__((ext_vector_type(4)))  unsigned v4u;
typedef __attribute__((ext_vector_type(8)))  int      v8i;
typedef __attribute__((ext_vector_type(4)))  int      v4i;

#if defined(__has_builtin)
#  if __has_builtin(__builtin_amdgcn_tensor_load_to_lds)
#    define HAVE_TDM 1
#  endif
#endif
#ifndef HAVE_TDM
#  define HAVE_TDM 0
#endif

#if HAVE_TDM
typedef const void __attribute__((address_space(3))) lds_cv;
__device__ static inline unsigned lds_off(const void* p) {
    // ptrtoint of an addrspace(3) pointer is the byte offset within the
    // workgroup's LDS allocation (what D#.lds_addr wants).
    return (unsigned)(unsigned long long)(lds_cv*)p;
}

// TDM 2D tile load: 32 (contiguous f32) x 64 rows, row stride = stride0 elems.
// D# packing per CDNA5 ISA ch.8 (group0: count/lds/global/type; group1:
// data_size=4B, tensor_dim == tile_dim (no OOB), dim0 stride).
__device__ static inline void tdm_load_tile_f32(unsigned ldsOff, const float* g,
                                                long stride0)
{
    const unsigned long long ga = (unsigned long long)g;
    v4u g0;
    g0[0] = 1u;                                                  // count=1, user D#
    g0[1] = ldsOff;                                              // lds_addr (bytes)
    g0[2] = (unsigned)ga;                                        // global_addr[31:0]
    g0[3] = (unsigned)((ga >> 32) & 0x01ffffffu) | 0x80000000u;  // [56:32] | type=2
    v8i g1;
    g1[0] = (int)(2u << 16);                                     // data_size=2 (4B), mask=0
    g1[1] = (int)(32u << 16);                                    // tensor_dim0[15:0]=32
    g1[2] = (int)(64u << 16);                                    // tensor_dim1[15:0]=64
    g1[3] = (int)(32u << 16);                                    // tile_dim0=32
    g1[4] = 64;                                                  // tile_dim1=64, tile_dim2=0
    g1[5] = (int)(unsigned)(stride0 & 0xffffffffu);              // dim0_stride[31:0]
    g1[6] = (int)(unsigned)((stride0 >> 32) & 0xffffu);          // dim0_stride[47:32]
    g1[7] = 0;
    const v4i gz = {0, 0, 0, 0};
#if __clang_major__ >= 23
    const v8i gz8 = {0, 0, 0, 0, 0, 0, 0, 0};
    __builtin_amdgcn_tensor_load_to_lds(g0, g1, gz, gz, gz8, 0);
#else
    __builtin_amdgcn_tensor_load_to_lds(g0, g1, gz, gz, 0);
#endif
}
#endif // HAVE_TDM

// VALU staging of a 64x32 panel (used for transposed operands / TDM fallback).
// T=0: dst[r][k] = src[(base+r)*lda + k0+k]; T=1: dst[r][k] = src[(k0+k)*lda + base+r]
template<bool T>
__device__ static inline void stage_panel(float* dst, const float* src, int lda,
                                          int base, int k0, int tid)
{
    const int row = tid >> 2;          // 0..63
    const int col = (tid & 3) * 8;     // 0,8,16,24
    #pragma unroll
    for (int i = 0; i < 8; ++i) {
        const int rr = base + row, kk = k0 + col + i;
        dst[row * 32 + col + i] = T ? src[(long)kk * lda + rr]
                                    : src[(long)rr * lda + kk];
    }
}

// Build a 16x32 f16 fragment from a f32 LDS row [32 k-values], converting.
// Lane (r,hi): k = hi*8 + 0..7 and 16 + hi*8 + 0..7  (ISA 16-bit A layout).
__device__ static inline v16h frag_f32(const float* rowBase, int hi)
{
    const v4f a0 = *(const v4f*)(rowBase + hi * 8);
    const v4f a1 = *(const v4f*)(rowBase + hi * 8 + 4);
    const v4f a2 = *(const v4f*)(rowBase + hi * 8 + 16);
    const v4f a3 = *(const v4f*)(rowBase + hi * 8 + 20);
    v16h f;
    #pragma unroll
    for (int i = 0; i < 4; ++i) {
        f[i]      = (_Float16)a0[i];
        f[4 + i]  = (_Float16)a1[i];
        f[8 + i]  = (_Float16)a2[i];
        f[12 + i] = (_Float16)a3[i];
    }
    return f;
}

// Batched GEMM: C[m,n] = alpha * sum_k Aop[m,k] * Bop[n,k]
//   TA/TB: operand read transposed (Aop[m,k]=A[k*lda+m], Bop[n,k]=B[k*ldb+n]).
// Batch z: A += z*strideA, B += (z%bMod)*strideB, C += z*strideC.
// Block: 256 threads / 8 waves; macro-tile 64(M) x 64(N); wave tile 16x32.
// Grid: x=N/64, y=M/64, z=batch. Requires M%64==0, N%64==0, K%32==0.
template<bool TA, bool TB>
__global__ __launch_bounds__(256)
void gemm_abt_wmma(const float* __restrict__ A, int lda, long strideA,
                   const float* __restrict__ Bm, int ldb, long strideB, int bMod,
                   float* __restrict__ C, int ldc, long strideC,
                   int K, float alpha)
{
    __shared__ __align__(16) float AsF[64 * 32];
    __shared__ __align__(16) float BsF[64 * 32];

    const int tid  = threadIdx.x;
    const int lane = tid & 31;
    const int wave = tid >> 5;
    const int wm   = wave >> 1;      // 0..3 (M sub-tile)
    const int wn   = wave & 1;       // 0..1 (N sub-tile, 32 wide)
    const int r    = lane & 15;
    const int hi   = lane >> 4;
    const int z    = blockIdx.z;
    const int m0   = blockIdx.y * 64;
    const int n0   = blockIdx.x * 64;

    A  += (long)z * strideA;
    Bm += (long)(z % bMod) * strideB;
    C  += (long)z * strideC;

#if HAVE_TDM
    const unsigned ldsA = lds_off(AsF);
    const unsigned ldsB = lds_off(BsF);
#endif

    v8f acc0 = {}, acc1 = {};

    for (int k0 = 0; k0 < K; k0 += 32) {
        if (TA || !HAVE_TDM) stage_panel<TA>(AsF, A, lda, m0, k0, tid);
        if (TB || !HAVE_TDM) stage_panel<TB>(BsF, Bm, ldb, n0, k0, tid);
#if HAVE_TDM
        if (!TA || !TB) {
            if (tid < 32) {                 // one wave drives the TDM
                if (!TA) tdm_load_tile_f32(ldsA, A + (long)m0 * lda + k0, lda);
                if (!TB) tdm_load_tile_f32(ldsB, Bm + (long)n0 * ldb + k0, ldb);
                __builtin_amdgcn_s_wait_tensorcnt(0);
            }
        }
#endif
        __syncthreads();

        const v16h af  = frag_f32(&AsF[(wm * 16 + r) * 32], hi);
        const v16h bf0 = frag_f32(&BsF[(wn * 32 + r) * 32], hi);
        const v16h bf1 = frag_f32(&BsF[(wn * 32 + 16 + r) * 32], hi);
        acc0 = __builtin_amdgcn_wmma_f32_16x16x32_f16(
                   false, af, false, bf0, (short)0, acc0, false, false);
        acc1 = __builtin_amdgcn_wmma_f32_16x16x32_f16(
                   false, af, false, bf1, (short)0, acc1, false, false);
        __syncthreads();
    }

    #pragma unroll
    for (int v = 0; v < 8; ++v) {
        const int mm = m0 + wm * 16 + v + 8 * hi;
        float* crow = C + (long)mm * ldc + n0 + wn * 32;
        crow[r]      = alpha * acc0[v];
        crow[16 + r] = alpha * acc1[v];
    }
}

// InstanceNorm stats: one block per (b,h) slice, biased variance over Cq*960.
__global__ __launch_bounds__(256)
void inorm_stats(const float* __restrict__ attn, float* __restrict__ stats,
                 long sliceElems)
{
    const int z = blockIdx.x;
    const float* p = attn + (long)z * sliceElems;
    float s = 0.f, ss = 0.f;
    for (long i = threadIdx.x; i < sliceElems; i += 256) {
        const float v = p[i];
        s += v; ss += v * v;
    }
    __shared__ float sh[256], sh2[256];
    sh[threadIdx.x] = s; sh2[threadIdx.x] = ss;
    __syncthreads();
    for (int w = 128; w > 0; w >>= 1) {
        if (threadIdx.x < w) {
            sh[threadIdx.x]  += sh[threadIdx.x + w];
            sh2[threadIdx.x] += sh2[threadIdx.x + w];
        }
        __syncthreads();
    }
    if (threadIdx.x == 0) {
        const float mean = sh[0] / (float)sliceElems;
        const float var  = sh2[0] / (float)sliceElems - mean * mean;
        stats[2 * z]     = mean;
        stats[2 * z + 1] = rsqrtf(var + 1e-5f);
    }
}

// Normalize with slice stats, then softmax over the 960-wide row.
__global__ __launch_bounds__(256)
void inorm_softmax_row(float* __restrict__ attn, const float* __restrict__ stats,
                       int Cq)
{
    const int row = blockIdx.x;             // z*Cq + c
    const int z   = row / Cq;
    const float mean = stats[2 * z];
    const float rstd = stats[2 * z + 1];
    float* p = attn + (long)row * 960;

    __shared__ float buf[960];
    __shared__ float red[256];

    float lmax = -3.0e38f;
    for (int i = threadIdx.x; i < 960; i += 256) {
        const float v = (p[i] - mean) * rstd;
        buf[i] = v;
        lmax = fmaxf(lmax, v);
    }
    red[threadIdx.x] = lmax; __syncthreads();
    for (int w = 128; w > 0; w >>= 1) {
        if (threadIdx.x < w) red[threadIdx.x] = fmaxf(red[threadIdx.x], red[threadIdx.x + w]);
        __syncthreads();
    }
    const float m = red[0];
    __syncthreads();

    float lsum = 0.f;
    for (int i = threadIdx.x; i < 960; i += 256) {
        const float e = __expf(buf[i] - m);
        buf[i] = e;
        lsum += e;
    }
    red[threadIdx.x] = lsum; __syncthreads();
    for (int w = 128; w > 0; w >>= 1) {
        if (threadIdx.x < w) red[threadIdx.x] += red[threadIdx.x + w];
        __syncthreads();
    }
    const float inv = 1.0f / red[0];
    for (int i = threadIdx.x; i < 960; i += 256) p[i] = buf[i] * inv;
}

extern "C" void kernel_launch(void* const* d_in, const int* in_sizes, int n_in,
                              void* d_out, int out_size, void* d_ws, size_t ws_size,
                              hipStream_t stream)
{
    (void)in_sizes; (void)n_in; (void)out_size; (void)ws_size;

    const float* Q[4]  = { (const float*)d_in[0], (const float*)d_in[1],
                           (const float*)d_in[2], (const float*)d_in[3] };
    const float* KV    =   (const float*)d_in[4];
    const float* Wk    =   (const float*)d_in[5];
    const float* Wv    =   (const float*)d_in[6];
    const float* Wq[4] = { (const float*)d_in[7], (const float*)d_in[8],
                           (const float*)d_in[9], (const float*)d_in[10] };
    const float* Wo[4] = { (const float*)d_in[11], (const float*)d_in[12],
                           (const float*)d_in[13], (const float*)d_in[14] };
    float* out = (float*)d_out;

    const int Z = 64;                 // B*H batches
    const long S = 960;
    const long NS = 128L * 960;       // per-batch [n, s] slab

    float* ws       = (float*)d_ws;
    float* KVt      = ws;                          // 64*128*960 f32
    float* attnBuf  = KVt + 64L * NS;              // 64*512*960 f32 (also Kt)
    float* QtBuf    = attnBuf + 64L * 512 * 960;   // 64*128*512 f32 (also outb)
    float* statsBuf = QtBuf + 64L * 128 * 512;     // 64*2 f32

    const dim3 blk(256);
    const float inv_scale = 1.0f / sqrtf(960.0f);

    // 1) Kt = KV · Wk^T  (Kt in attnBuf)           [z: 128 x 960, K=960]
    gemm_abt_wmma<false, false><<<dim3(15, 2, Z), blk, 0, stream>>>(
        KV, 960, NS, Wk, 960, S * S, 8, attnBuf, 960, NS, 960, 1.0f);

    // 2) KVt = Kt · Wv^T                           [z: 128 x 960, K=960]
    gemm_abt_wmma<false, false><<<dim3(15, 2, Z), blk, 0, stream>>>(
        attnBuf, 960, NS, Wv, 960, S * S, 8, KVt, 960, NS, 960, 1.0f);

    long outOff = 0;
    for (int br = 0; br < 4; ++br) {
        const int Cq = 64 << br;
        const long CC = (long)Cq * Cq;

        // a) Qt = Q · Wq^T                         [z: 128 x Cq, K=Cq]
        gemm_abt_wmma<false, false><<<dim3(Cq / 64, 2, Z), blk, 0, stream>>>(
            Q[br], Cq, 128L * Cq, Wq[br], Cq, CC, 8,
            QtBuf, Cq, 128L * Cq, Cq, 1.0f);

        // b) attn = (Qt^T · KVt) * inv_scale       [z: Cq x 960, K=128]
        gemm_abt_wmma<true, true><<<dim3(15, Cq / 64, Z), blk, 0, stream>>>(
            QtBuf, Cq, 128L * Cq, KVt, 960, NS, Z,
            attnBuf, 960, (long)Cq * 960, 128, inv_scale);

        // c) InstanceNorm stats per (b,h) slice
        inorm_stats<<<Z, blk, 0, stream>>>(attnBuf, statsBuf, (long)Cq * 960);

        // d) normalize + softmax over s
        inorm_softmax_row<<<Z * Cq, blk, 0, stream>>>(attnBuf, statsBuf, Cq);

        // e) outb = attn · KVt^T  (into QtBuf)     [z: Cq x 128, K=960]
        gemm_abt_wmma<false, false><<<dim3(2, Cq / 64, Z), blk, 0, stream>>>(
            attnBuf, 960, (long)Cq * 960, KVt, 960, NS, Z,
            QtBuf, 128, (long)Cq * 128, 960, 1.0f);

        // f) final = outb^T · Wo^T -> d_out        [z: 128 x Cq, K=Cq]
        gemm_abt_wmma<true, false><<<dim3(Cq / 64, 2, Z), blk, 0, stream>>>(
            QtBuf, 128, (long)Cq * 128, Wo[br], Cq, 0L, 1,
            out + outOff, Cq, 128L * Cq, Cq, 1.0f);

        outOff += 8192L * Cq;   // B*N*Cq elements per branch
    }
}
